// BuildingPointNet_19679540150375
// MI455X (gfx1250) — matrix operand order repaired and checked by reading, not compiled, not run on hardware
//
#include <hip/hip_runtime.h>
#include <hip/hip_bf16.h>

// ---------------------------------------------------------------------------
// BuildingPointNet on MI455X (gfx1250): bf16 WMMA pipeline.
//   normals -> conv1 (9->64->128, max over 17 nbrs) -> conv2 (131->128->256)
//   -> classifier (256->8) -> log_softmax
// One wave32 per 16-point block; message MLPs run as v_wmma_f32_16x16x32_bf16
// tile chains with A/activations staged through LDS and weights pre-packed
// into B-fragment layout in the workspace.
// ---------------------------------------------------------------------------

#define NPTS 40000
#define KNBR 16
#define BN_EPS 1e-5f

typedef __bf16 bf16;
typedef bf16  v16bf __attribute__((ext_vector_type(16)));
typedef float v8f   __attribute__((ext_vector_type(8)));
typedef int   v4i   __attribute__((ext_vector_type(4)));

union B8U   { v4i v; bf16 b[8]; };
union FragU { struct { v4i a, b; } s; v16bf f; };
union BfBits { bf16 h; unsigned short s; };

static __device__ inline v4i zero4() { v4i z = {0, 0, 0, 0}; return z; }
static __device__ inline v8f splat8(float x) { v8f r = {x, x, x, x, x, x, x, x}; return r; }

static __device__ inline void dsfence() {
  asm volatile("s_wait_dscnt 0" ::: "memory");
}

// A fragment (16x32 bf16): lanes 0-15 carry row M=lane (K 0..15),
// lanes 16-31 carry row M=lane-16 (K 16..31). Memory is row-major.
static __device__ inline v16bf ldsA(const char* buf, int rowBytes, int kOff, int lane) {
  int nl = lane & 15, h = (lane >> 4) & 1;
  const char* p = buf + nl * rowBytes + kOff + h * 32;
  FragU u;
  u.s.a = *(const v4i*)p;
  u.s.b = *(const v4i*)(p + 16);
  return u.f;
}

// B fragment (32x16 bf16): packed so each lane reads 8 consecutive dwords.
static __device__ inline v16bf ldsB(const char* tile, int lane) {
  const char* p = tile + lane * 32;
  FragU u;
  u.s.a = *(const v4i*)p;
  u.s.b = *(const v4i*)(p + 16);
  return u.f;
}

static __device__ inline v8f wmma_bf16(v16bf a, v16bf b, v8f c) {
  return __builtin_amdgcn_wmma_f32_16x16x32_bf16(false, a, false, b, (short)0, c, false, false);
}

// Store a C-layout f32 tile as bf16 into a row-major LDS matrix.
static __device__ inline void stHidTile(char* buf, int rowBytes, int colBase, v8f t, int lane) {
  int nl = lane & 15, h = (lane >> 4) & 1;
  int colB = (colBase + nl) * 2;
#pragma unroll
  for (int v = 0; v < 8; ++v) {
    *(bf16*)(buf + (v + 8 * h) * rowBytes + colB) = (bf16)t[v];
  }
}

// ---------------------------------------------------------------------------
// Pack f32 weight [K x N] into bf16 B-fragment tiles: tile t = kt*NT + nt,
// dword (t*256 + lane*8 + v) holds K = kt*32 + (lane>=16 ? 16:0) + 2v (+1 in
// high half), N = nt*16 + (lane&15). Out-of-range -> 0 (zero K padding).
// ---------------------------------------------------------------------------
__global__ void pack_weight_kernel(const float* __restrict__ W, int K, int N,
                                   int KT, int NT, unsigned* __restrict__ dst) {
  int tid = blockIdx.x * blockDim.x + threadIdx.x;
  int total = KT * NT * 256;
  if (tid >= total) return;
  int v = tid & 7;
  int lane = (tid >> 3) & 31;
  int t = tid >> 8;
  int nt = t % NT, kt = t / NT;
  int n = nt * 16 + (lane & 15);
  int k0 = kt * 32 + ((lane >> 4) ? 16 : 0) + 2 * v;
  float lo = (k0 < K && n < N) ? W[k0 * N + n] : 0.0f;
  float hi = (k0 + 1 < K && n < N) ? W[(k0 + 1) * N + n] : 0.0f;
  BfBits a, b;
  a.h = (bf16)lo;
  b.h = (bf16)hi;
  dst[tid] = (unsigned)a.s | ((unsigned)b.s << 16);
}

// ---------------------------------------------------------------------------
// Normals: cross of first two neighbor offsets, normalized, fallback [0,0,1].
// ---------------------------------------------------------------------------
__global__ void normals_kernel(const float* __restrict__ pos, const int* __restrict__ nbr,
                               float* __restrict__ out) {
  int i = blockIdx.x * blockDim.x + threadIdx.x;
  if (i >= NPTS) return;
  int j0 = nbr[i * KNBR + 0], j1 = nbr[i * KNBR + 1];
  float px = pos[3 * i], py = pos[3 * i + 1], pz = pos[3 * i + 2];
  float ax = pos[3 * j0] - px, ay = pos[3 * j0 + 1] - py, az = pos[3 * j0 + 2] - pz;
  float bx = pos[3 * j1] - px, by = pos[3 * j1 + 1] - py, bz = pos[3 * j1 + 2] - pz;
  float cx = ay * bz - az * by;
  float cy = az * bx - ax * bz;
  float cz = ax * by - ay * bx;
  float nn = sqrtf(cx * cx + cy * cy + cz * cz);
  float inv = 1.0f / fmaxf(nn, 1e-12f);
  if (nn > 0.0f) {
    out[3 * i] = cx * inv; out[3 * i + 1] = cy * inv; out[3 * i + 2] = cz * inv;
  } else {
    out[3 * i] = 0.0f; out[3 * i + 1] = 0.0f; out[3 * i + 2] = 1.0f;
  }
}

// ---------------------------------------------------------------------------
// Layer 1: per 16-point block, 17 slots of (9->64 relu BN ->128), max-agg,
// relu, store x1 as bf16 [N x 128]. 4 waves per workgroup.
// ---------------------------------------------------------------------------
__global__ __launch_bounds__(128) void layer1_kernel(
    const float* __restrict__ pos, const int* __restrict__ nbr, const float* __restrict__ nrm,
    const float* __restrict__ b1, const float* __restrict__ g1, const float* __restrict__ bt1,
    const float* __restrict__ m1, const float* __restrict__ v1, const float* __restrict__ b2,
    const unsigned* __restrict__ w1p, const unsigned* __restrict__ w2p,
    bf16* __restrict__ x1out) {
  extern __shared__ char smem[];
  unsigned* w1l = (unsigned*)smem;   // 1024 dw = 4KB   (1 kstep x 4 ntiles)
  unsigned* w2l = w1l + 1024;        // 4096 dw = 16KB  (2 ksteps x 8 ntiles)
  char* arenas = (char*)(w2l + 4096);
  const int tid = threadIdx.x;
  for (int i = tid; i < 1024; i += 128) w1l[i] = w1p[i];
  for (int i = tid; i < 4096; i += 128) w2l[i] = w2p[i];
  __syncthreads();

  const int wave = tid >> 5, lane = tid & 31;
  const int nl = lane & 15, hh = lane >> 4;
  char* rowbuf = arenas + wave * 3072;  // 16 rows x 64B  (K=32 bf16)
  char* hid = rowbuf + 1024;            // 16 rows x 128B (64 bf16)
  const int base = (blockIdx.x * 4 + wave) * 16;

  float sc1[4], sh1[4], bb1[4];
#pragma unroll
  for (int t = 0; t < 4; ++t) {
    int c = t * 16 + nl;
    float s = g1[c] * rsqrtf(v1[c] + BN_EPS);
    sc1[t] = s;
    sh1[t] = bt1[c] - m1[c] * s;
    bb1[t] = b1[c];
  }
  float bb2[8];
#pragma unroll
  for (int t = 0; t < 8; ++t) bb2[t] = b2[t * 16 + nl];

  const int p = base + nl;
  const float pix = pos[3 * p], piy = pos[3 * p + 1], piz = pos[3 * p + 2];

  v8f acc[8];
#pragma unroll
  for (int t = 0; t < 8; ++t) acc[t] = splat8(-3.0e38f);

  for (int slot = 0; slot < 17; ++slot) {
    if (!hh) {  // lanes 0-15 each build one 9-feature row (zero padded to 32)
      int j = (slot < KNBR) ? nbr[p * KNBR + slot] : p;
      float jx = pos[3 * j], jy = pos[3 * j + 1], jz = pos[3 * j + 2];
      B8U u0, u1;
      u0.b[0] = (bf16)jx; u0.b[1] = (bf16)jy; u0.b[2] = (bf16)jz;
      u0.b[3] = (bf16)nrm[3 * j]; u0.b[4] = (bf16)nrm[3 * j + 1]; u0.b[5] = (bf16)nrm[3 * j + 2];
      u0.b[6] = (bf16)(jx - pix); u0.b[7] = (bf16)(jy - piy);
      u1.v = zero4();
      u1.b[0] = (bf16)(jz - piz);
      v4i* r = (v4i*)(rowbuf + nl * 64);
      r[0] = u0.v; r[1] = u1.v; r[2] = zero4(); r[3] = zero4();
    }
    dsfence();
    v16bf a1 = ldsA(rowbuf, 64, 0, lane);
#pragma unroll
    for (int t = 0; t < 4; ++t) {  // 9(->32) x 64, relu + BN, stage bf16
      v8f c = splat8(bb1[t]);
      c = wmma_bf16(a1, ldsB((const char*)w1l + t * 1024, lane), c);
#pragma unroll
      for (int v = 0; v < 8; ++v) {
        float x = fmaxf(c[v], 0.0f);
        c[v] = x * sc1[t] + sh1[t];
      }
      stHidTile(hid, 128, t * 16, c, lane);
    }
    dsfence();
    v16bf a20 = ldsA(hid, 128, 0, lane);
    v16bf a21 = ldsA(hid, 128, 64, lane);
#pragma unroll
    for (int t = 0; t < 8; ++t) {  // 64 x 128, max-aggregate
      v8f c = splat8(bb2[t]);
      c = wmma_bf16(a20, ldsB((const char*)w2l + t * 1024, lane), c);
      c = wmma_bf16(a21, ldsB((const char*)w2l + (8 + t) * 1024, lane), c);
#pragma unroll
      for (int v = 0; v < 8; ++v) acc[t][v] = fmaxf(acc[t][v], c[v]);
    }
  }
#pragma unroll
  for (int t = 0; t < 8; ++t) {
    int col = t * 16 + nl;
#pragma unroll
    for (int v = 0; v < 8; ++v) {
      int m = v + 8 * hh;
      x1out[(size_t)(base + m) * 128 + col] = (bf16)fmaxf(acc[t][v], 0.0f);
    }
  }
}

// ---------------------------------------------------------------------------
// Layer 2 + classifier: per 16-point block, 17 slots of (131->128 relu BN
// ->256), max-agg, relu, then 256->8 WMMA classifier + log_softmax.
// 2 waves per workgroup (LDS: 112KB weights + 2x9KB staging).
// ---------------------------------------------------------------------------
__global__ __launch_bounds__(64) void layer2_kernel(
    const float* __restrict__ pos, const int* __restrict__ nbr, const bf16* __restrict__ x1,
    const float* __restrict__ b3, const float* __restrict__ g2, const float* __restrict__ bt2,
    const float* __restrict__ m2, const float* __restrict__ v2, const float* __restrict__ b4,
    const float* __restrict__ bc,
    const unsigned* __restrict__ w3p, const unsigned* __restrict__ w4p,
    const unsigned* __restrict__ wcp, float* __restrict__ out) {
  extern __shared__ char smem[];
  unsigned* w3l = (unsigned*)smem;   // 10240 dw (5 ksteps x 8 ntiles)
  unsigned* w4l = w3l + 10240;       // 16384 dw (4 ksteps x 16 ntiles)
  unsigned* wcl = w4l + 16384;       // 2048 dw  (8 ksteps x 1 ntile)
  char* arenas = (char*)(wcl + 2048);
  const int tid = threadIdx.x;
  for (int i = tid; i < 10240; i += 64) w3l[i] = w3p[i];
  for (int i = tid; i < 16384; i += 64) w4l[i] = w4p[i];
  for (int i = tid; i < 2048; i += 64) wcl[i] = wcp[i];
  __syncthreads();

  const int wave = tid >> 5, lane = tid & 31;
  const int nl = lane & 15, hh = lane >> 4;
  char* rowbuf = arenas + wave * 9216;  // 16 rows x 320B (K=160 bf16)
  char* hid = rowbuf + 5120;            // 16 rows x 256B (128 bf16)
  const int base = (blockIdx.x * 2 + wave) * 16;

  float sc2[8], sh2[8], bb3[8];
#pragma unroll
  for (int t = 0; t < 8; ++t) {
    int c = t * 16 + nl;
    float s = g2[c] * rsqrtf(v2[c] + BN_EPS);
    sc2[t] = s;
    sh2[t] = bt2[c] - m2[c] * s;
    bb3[t] = b3[c];
  }
  float bb4[16];
#pragma unroll
  for (int t = 0; t < 16; ++t) bb4[t] = b4[t * 16 + nl];
  const float bcl = (nl < 8) ? bc[nl] : 0.0f;

  const int p = base + nl;
  const float pix = pos[3 * p], piy = pos[3 * p + 1], piz = pos[3 * p + 2];

  v8f acc[16];
#pragma unroll
  for (int t = 0; t < 16; ++t) acc[t] = splat8(-3.0e38f);

  for (int slot = 0; slot < 17; ++slot) {
    int j = (slot < KNBR) ? nbr[p * KNBR + slot] : p;
    {  // gather x1[j] (256B) split across the two lane halves; high half
       // also writes the [rel | zero-pad] tail (cols 128..159).
      const v4i* src = (const v4i*)(x1 + (size_t)j * 128);
      v4i* dr = (v4i*)(rowbuf + nl * 320);
#pragma unroll
      for (int q = 0; q < 8; ++q) dr[hh * 8 + q] = src[hh * 8 + q];
      if (hh) {
        B8U u;
        u.v = zero4();
        u.b[0] = (bf16)(pos[3 * j] - pix);
        u.b[1] = (bf16)(pos[3 * j + 1] - piy);
        u.b[2] = (bf16)(pos[3 * j + 2] - piz);
        dr[16] = u.v; dr[17] = zero4(); dr[18] = zero4(); dr[19] = zero4();
      }
      if (slot + 1 < 17) {  // warm next slot's gather
        int jn = (slot + 1 < KNBR) ? nbr[p * KNBR + slot + 1] : p;
        __builtin_prefetch(x1 + (size_t)jn * 128, 0, 0);
      }
    }
    dsfence();
    v16bf a[5];
#pragma unroll
    for (int ks = 0; ks < 5; ++ks) a[ks] = ldsA(rowbuf, 320, ks * 64, lane);
#pragma unroll
    for (int t = 0; t < 8; ++t) {  // 131(->160) x 128, relu + BN, stage bf16
      v8f c = splat8(bb3[t]);
#pragma unroll
      for (int ks = 0; ks < 5; ++ks)
        c = wmma_bf16(a[ks], ldsB((const char*)w3l + (ks * 8 + t) * 1024, lane), c);
#pragma unroll
      for (int v = 0; v < 8; ++v) {
        float x = fmaxf(c[v], 0.0f);
        c[v] = x * sc2[t] + sh2[t];
      }
      stHidTile(hid, 256, t * 16, c, lane);
    }
    dsfence();
    v16bf a2[4];
#pragma unroll
    for (int ks = 0; ks < 4; ++ks) a2[ks] = ldsA(hid, 256, ks * 64, lane);
#pragma unroll
    for (int t = 0; t < 16; ++t) {  // 128 x 256, max-aggregate
      v8f c = splat8(bb4[t]);
#pragma unroll
      for (int ks = 0; ks < 4; ++ks)
        c = wmma_bf16(a2[ks], ldsB((const char*)w4l + (ks * 16 + t) * 1024, lane), c);
#pragma unroll
      for (int v = 0; v < 8; ++v) acc[t][v] = fmaxf(acc[t][v], c[v]);
    }
  }

  // Classifier: relu(x2) staged bf16 (16 x 512B rows), then 8 WMMAs.
  char* xs = rowbuf;  // staging buffers are dead; reuse 8KB of the arena
#pragma unroll
  for (int t = 0; t < 16; ++t) {
    v8f c;
#pragma unroll
    for (int v = 0; v < 8; ++v) c[v] = fmaxf(acc[t][v], 0.0f);
    stHidTile(xs, 512, t * 16, c, lane);
  }
  dsfence();
  v8f lg = splat8(bcl);
#pragma unroll
  for (int ks = 0; ks < 8; ++ks) {
    v16bf ax = ldsA(xs, 512, ks * 64, lane);
    lg = wmma_bf16(ax, ldsB((const char*)wcl + ks * 1024, lane), lg);
  }
  // log_softmax over the 8 class lanes (xor<8 stays inside each lane group).
#pragma unroll
  for (int v = 0; v < 8; ++v) {
    float x = lg[v];
    float mx = x;
    mx = fmaxf(mx, __shfl_xor(mx, 1, 32));
    mx = fmaxf(mx, __shfl_xor(mx, 2, 32));
    mx = fmaxf(mx, __shfl_xor(mx, 4, 32));
    float e = __expf(x - mx);
    float s = e;
    s += __shfl_xor(s, 1, 32);
    s += __shfl_xor(s, 2, 32);
    s += __shfl_xor(s, 4, 32);
    float ls = (x - mx) - __logf(s);
    if (nl < 8) out[(size_t)(base + v + 8 * hh) * 8 + nl] = ls;
  }
}

// ---------------------------------------------------------------------------
// Workspace layout (bytes)
// ---------------------------------------------------------------------------
static const size_t OFF_NORM = 0;                         // 40000*3*4 = 480000
static const size_t OFF_X1   = 480256;                    // 40000*128*2 = 10240000
static const size_t OFF_W1P  = OFF_X1 + 10240000;         // 1*4 tiles  * 1KB = 4096
static const size_t OFF_W2P  = OFF_W1P + 4096;            // 2*8 tiles  = 16384
static const size_t OFF_W3P  = OFF_W2P + 16384;           // 5*8 tiles  = 40960
static const size_t OFF_W4P  = OFF_W3P + 40960;           // 4*16 tiles = 65536
static const size_t OFF_WCP  = OFF_W4P + 65536;           // 8*1 tiles  = 8192

extern "C" void kernel_launch(void* const* d_in, const int* in_sizes, int n_in,
                              void* d_out, int out_size, void* d_ws, size_t ws_size,
                              hipStream_t stream) {
  (void)in_sizes; (void)n_in; (void)out_size; (void)ws_size;
  const float* pos = (const float*)d_in[0];
  const int*   nbr = (const int*)d_in[1];
  const float* W1  = (const float*)d_in[2];
  const float* b1  = (const float*)d_in[3];
  const float* g1  = (const float*)d_in[4];
  const float* bt1 = (const float*)d_in[5];
  const float* m1  = (const float*)d_in[6];
  const float* v1  = (const float*)d_in[7];
  const float* W2  = (const float*)d_in[8];
  const float* b2  = (const float*)d_in[9];
  const float* W3  = (const float*)d_in[10];
  const float* b3  = (const float*)d_in[11];
  const float* g2  = (const float*)d_in[12];
  const float* bt2 = (const float*)d_in[13];
  const float* m2  = (const float*)d_in[14];
  const float* v2  = (const float*)d_in[15];
  const float* W4  = (const float*)d_in[16];
  const float* b4  = (const float*)d_in[17];
  const float* Wc  = (const float*)d_in[18];
  const float* bc  = (const float*)d_in[19];

  char* ws = (char*)d_ws;
  float*    normals = (float*)(ws + OFF_NORM);
  bf16*     x1      = (bf16*)(ws + OFF_X1);
  unsigned* w1p     = (unsigned*)(ws + OFF_W1P);
  unsigned* w2p     = (unsigned*)(ws + OFF_W2P);
  unsigned* w3p     = (unsigned*)(ws + OFF_W3P);
  unsigned* w4p     = (unsigned*)(ws + OFF_W4P);
  unsigned* wcp     = (unsigned*)(ws + OFF_WCP);

  // Pack weights into bf16 B-fragment tiles (zero-padded K/N).
  pack_weight_kernel<<<(1 * 4 * 256 + 255) / 256, 256, 0, stream>>>(W1, 9, 64, 1, 4, w1p);
  pack_weight_kernel<<<(2 * 8 * 256 + 255) / 256, 256, 0, stream>>>(W2, 64, 128, 2, 8, w2p);
  pack_weight_kernel<<<(5 * 8 * 256 + 255) / 256, 256, 0, stream>>>(W3, 131, 128, 5, 8, w3p);
  pack_weight_kernel<<<(4 * 16 * 256 + 255) / 256, 256, 0, stream>>>(W4, 128, 256, 4, 16, w4p);
  pack_weight_kernel<<<(8 * 1 * 256 + 255) / 256, 256, 0, stream>>>(Wc, 256, 8, 8, 1, wcp);

  normals_kernel<<<(NPTS + 255) / 256, 256, 0, stream>>>(pos, nbr, normals);

  // 2500 blocks of 16 points; layer1: 4 waves/WG, layer2: 2 waves/WG.
  size_t lds1 = 4096 + 16384 + 4 * 3072;                 // 32768 B
  layer1_kernel<<<625, 128, lds1, stream>>>(pos, nbr, normals,
                                            b1, g1, bt1, m1, v1, b2,
                                            w1p, w2p, x1);

  size_t lds2 = (10240 + 16384 + 2048) * 4 + 2 * 9216;   // 133120 B
  layer2_kernel<<<1250, 64, lds2, stream>>>(pos, nbr, x1,
                                            b3, g2, bt2, m2, v2, b4, bc,
                                            w3p, w4p, wcp, (float*)d_out);
}
